// TriEinsumAttention_86122684220184
// MI455X (gfx1250) — matrix operand order repaired and checked by reading, not compiled, not run on hardware
//
#include <hip/hip_runtime.h>
#include <hip/hip_bf16.h>
#include <cstdint>

typedef __bf16 bf16_t;
typedef bf16_t v16bf __attribute__((ext_vector_type(16)));
typedef bf16_t v8bf  __attribute__((ext_vector_type(8)));
typedef float  v8f   __attribute__((ext_vector_type(8)));
typedef unsigned int u32x4 __attribute__((ext_vector_type(4)));
typedef int i32x4 __attribute__((ext_vector_type(4)));
typedef int i32x8 __attribute__((ext_vector_type(8)));

#define B_  4
#define LA  256
#define LB  256
#define LC  32
#define DM  512
#define NH  8
#define DH  64

static __device__ __forceinline__ v8f wmma_bf16(v16bf a, v16bf b, v8f c) {
  // D = A(16x32 bf16) x B(32x16 bf16) + C(16x16 f32)
  return __builtin_amdgcn_wmma_f32_16x16x32_bf16(false, a, false, b, (short)0, c,
                                                 false, false);
}

static __device__ __forceinline__ v8f vzero8() {
  v8f z;
#pragma unroll
  for (int i = 0; i < 8; ++i) z[i] = 0.f;
  return z;
}

// Build a 16-element bf16 fragment from two contiguous 16-byte chunks.
static __device__ __forceinline__ v16bf ld16(const bf16_t* p0, const bf16_t* p1) {
  v8bf a = *(const v8bf*)p0;
  v8bf b = *(const v8bf*)p1;
  return __builtin_shufflevector(a, b, 0, 1, 2, 3, 4, 5, 6, 7,
                                 8, 9, 10, 11, 12, 13, 14, 15);
}
static __device__ __forceinline__ v8bf lo8(v16bf v) {
  return __builtin_shufflevector(v, v, 0, 1, 2, 3, 4, 5, 6, 7);
}
static __device__ __forceinline__ v8bf hi8(v16bf v) {
  return __builtin_shufflevector(v, v, 8, 9, 10, 11, 12, 13, 14, 15);
}

// ---- Tensor Data Mover: 2D tile (== whole tensor) global -> LDS -------------
// D# packing per CDNA5 ISA ch.8: group0 {count, lds_addr, global_addr, type=2},
// group1 {data_size=2B, tensor_dim0/1, tile_dim0/1, dim0_stride}.
#if __has_builtin(__builtin_amdgcn_tensor_load_to_lds)
#define HAVE_TDM 1
static __device__ __forceinline__ void tdm_load_2d(uint32_t lds_addr,
                                                   const void* gptr,
                                                   uint32_t dim0, uint32_t dim1,
                                                   uint32_t stride0) {
  uint64_t ga = (uint64_t)(uintptr_t)gptr;
  u32x4 g0;
  g0[0] = 1u;                                   // count=1, user descriptor
  g0[1] = lds_addr;                             // LDS byte address
  g0[2] = (uint32_t)ga;                         // global_addr[31:0]
  g0[3] = (uint32_t)((ga >> 32) & 0x01ffffffu) | (2u << 30);  // [56:32]|type=2
  i32x8 g1;
  g1[0] = (int)(1u << 16);                      // data_size=1 -> 2 bytes/elt
  g1[1] = (int)((dim0 & 0xffffu) << 16);        // tensor_dim0[15:0] @ bit48
  g1[2] = (int)(((dim0 >> 16) & 0xffffu) | ((dim1 & 0xffffu) << 16));
  g1[3] = (int)(((dim1 >> 16) & 0xffffu) | ((dim0 & 0xffffu) << 16)); // tile_dim0
  g1[4] = (int)(dim1 & 0xffffu);                // tile_dim1 (tile_dim2=0)
  g1[5] = (int)stride0;                         // tensor_dim0_stride[31:0]
  g1[6] = 0;
  g1[7] = 0;
  i32x4 z4;
  z4[0] = 0; z4[1] = 0; z4[2] = 0; z4[3] = 0;
#if defined(__clang_major__) && (__clang_major__ >= 23)
  i32x8 z8;
#pragma unroll
  for (int i = 0; i < 8; ++i) z8[i] = 0;
  __builtin_amdgcn_tensor_load_to_lds(g0, g1, z4, z4, z8, 0);
#else
  __builtin_amdgcn_tensor_load_to_lds(g0, g1, z4, z4, 0);
#endif
}
#else
#define HAVE_TDM 0
#endif

// ---------------------------------------------------------------- converts
__global__ void k_cvt_bf16(const float* __restrict__ src, bf16_t* __restrict__ dst,
                           int n) {
  for (int i = blockIdx.x * blockDim.x + threadIdx.x; i < n;
       i += gridDim.x * blockDim.x)
    dst[i] = (bf16_t)src[i];
}

// C_tokens [B,Lc,512] f32 -> Ch [B,h,Lc,64] bf16 (head split)
__global__ void k_pack_c(const float* __restrict__ ctok, bf16_t* __restrict__ ch) {
  int i = blockIdx.x * blockDim.x + threadIdx.x;
  if (i >= B_ * NH * LC * DH) return;
  int d = i & 63;
  int c = (i >> 6) & 31;
  int hh = (i >> 11) & 7;
  int b = i >> 14;
  ch[i] = (bf16_t)ctok[(b * LC + c) * DM + hh * DH + d];
}

// ------------------------------------------------------- projection GEMMs
// y[m,n] = sum_k x[m,k] * W[n,k] + bias[n]; M=1024, N=512, K=512.
// sel 0: Q -> Qh[b,h,l,d] ; sel 1: K -> Kh[b,h,l,d] ; sel 2: V -> Vt[b,h,d,l]
__global__ void __launch_bounds__(32) k_proj(
    const bf16_t* __restrict__ xq, const bf16_t* __restrict__ xk,
    const bf16_t* __restrict__ xv, const bf16_t* __restrict__ wq,
    const bf16_t* __restrict__ wk, const bf16_t* __restrict__ wv,
    const float* __restrict__ bq, const float* __restrict__ bk,
    const float* __restrict__ bv, bf16_t* __restrict__ qh,
    bf16_t* __restrict__ kh, bf16_t* __restrict__ vt) {
  const int lane = threadIdx.x & 31;
  const int l4 = lane & 15;
  const int hi = lane >> 4;  // 0/1
  const int sel = blockIdx.z;
  const bf16_t* X = sel == 0 ? xq : (sel == 1 ? xk : xv);
  const bf16_t* W = sel == 0 ? wq : (sel == 1 ? wk : wv);
  const float* Bv = sel == 0 ? bq : (sel == 1 ? bk : bv);
  const int mbase = blockIdx.x * 16;
  const int nbase = blockIdx.y * 16;

  const bf16_t* xrow = X + (size_t)(mbase + l4) * DM;  // A: row = m (lane&15)
  const bf16_t* wrow = W + (size_t)(nbase + l4) * DM;  // B: col = n (lane&15)
  const int a_off = hi * 8;
  const int b_off = hi * 8;

  v8f acc = vzero8();
  for (int kk = 0; kk < DM; kk += 32) {
    v16bf A = ld16(xrow + kk + a_off, xrow + kk + 16 + a_off);
    v16bf Bm = ld16(wrow + kk + b_off, wrow + kk + b_off + 16);
    acc = wmma_bf16(A, Bm, acc);
  }
  const int n = nbase + l4;
  const float bias = Bv[n];
  const int hsel = n >> 6;         // head
  const int d = n & 63;            // within-head dim
  const int bidx = mbase >> 8;     // batch (tiles never straddle a batch)
  const int lloc = (mbase & 255) + hi * 8;

  if (sel < 2) {
    bf16_t* dst = (sel == 0 ? qh : kh);
#pragma unroll
    for (int r = 0; r < 8; ++r)
      dst[((size_t)(bidx * NH + hsel) * LB + lloc + r) * DH + d] =
          (bf16_t)(acc[r] + bias);
  } else {
    // transposed store: lane holds fixed d, 8 consecutive l -> one b128 store
    v8bf o;
#pragma unroll
    for (int r = 0; r < 8; ++r) o[r] = (bf16_t)(acc[r] + bias);
    *(v8bf*)(vt + ((size_t)(bidx * NH + hsel) * DH + d) * LB + lloc) = o;
  }
}

// ------------------------------------------------- fused tri-linear attention
// One wave per (b,h, 16-row q tile).
__global__ void __launch_bounds__(32) k_tri_attn(
    const bf16_t* __restrict__ qh, const bf16_t* __restrict__ kh,
    const bf16_t* __restrict__ vt, const bf16_t* __restrict__ ch,
    float* __restrict__ outp, float* __restrict__ attnp) {
  __shared__ bf16_t sK[LB * DH];       // 32 KB   K row-major [k][d]
  __shared__ bf16_t sC[LC * DH];       // 4 KB    C row-major [c][d]
  __shared__ bf16_t sQc[LC * 1024];    // 64 KB   per-c Qc A-fragments (per lane)
  __shared__ bf16_t sAb[16 * 32];      // 1 KB    attn D->A transpose staging
  __shared__ float sA32[16 * 512];     // 32 KB   attn coalescing buffer

  const int lane = threadIdx.x & 31;
  const int l4 = lane & 15;
  const int hi = lane >> 4;
  const int bh = blockIdx.y;  // b*8+h
  const int b = bh >> 3, h = bh & 7;
  const int qbase = blockIdx.x * 16;
  const int a_off = hi * 8;
  const int b_off = hi * 8;

  const bf16_t* gk = kh + (size_t)bh * LB * DH;
  const bf16_t* gc = ch + (size_t)bh * LC * DH;
  const bf16_t* gvt = vt + (size_t)bh * DH * LB;  // V^T [d][k], read in pass 2b

  // ---- stage K, C into LDS via the Tensor Data Mover ----
#if HAVE_TDM
  tdm_load_2d((uint32_t)(uintptr_t)&sK[0], gk, DH, LB, DH);  // 256x64 bf16 tile
  tdm_load_2d((uint32_t)(uintptr_t)&sC[0], gc, DH, LC, DH);  // 32x64 bf16 tile
  // warm L2/WGP$ with V^T while the TDM fills LDS
  for (int i = lane; i < 256; i += 32)
    __builtin_prefetch(gvt + (size_t)i * 64, 0, 0);
  __builtin_amdgcn_s_wait_tensorcnt(0);
#else
  for (int i = lane; i < (LB * DH) / 8; i += 32)
    *(v8bf*)&sK[i * 8] = *(const v8bf*)(gk + i * 8);
  for (int i = lane; i < (LC * DH) / 8; i += 32)
    *(v8bf*)&sC[i * 8] = *(const v8bf*)(gc + i * 8);
  for (int i = lane; i < 256; i += 32)
    __builtin_prefetch(gvt + (size_t)i * 64, 0, 0);
#endif
  __syncthreads();

  // ---- Q fragments (A layout), row = qbase + (lane&15) ----
  const bf16_t* qrow = qh + ((size_t)bh * LA + qbase + l4) * DH;
  v16bf qf0 = ld16(qrow + a_off, qrow + 16 + a_off);
  v16bf qf1 = ld16(qrow + 32 + a_off, qrow + 48 + a_off);

  float m[8], s[8];
#pragma unroll
  for (int r = 0; r < 8; ++r) {
    m[r] = -3.0e38f;
    s[r] = 0.f;
  }

  // ================= pass 1: Qc precompute + online softmax stats ==========
  for (int c = 0; c < LC; ++c) {
    const bf16_t* crow = sC + c * DH;
    v16bf cf0 = ld16(crow + a_off, crow + 16 + a_off);
    v16bf cf1 = ld16(crow + 32 + a_off, crow + 48 + a_off);
    v16bf qc0, qc1;
#pragma unroll
    for (int j = 0; j < 16; ++j) {
      qc0[j] = (bf16_t)((float)qf0[j] * (float)cf0[j] * 0.125f);  // 1/sqrt(64)
      qc1[j] = (bf16_t)((float)qf1[j] * (float)cf1[j] * 0.125f);
    }
    bf16_t* qcs = sQc + c * 1024 + lane * 32;
    *(v8bf*)(qcs + 0) = lo8(qc0);
    *(v8bf*)(qcs + 8) = hi8(qc0);
    *(v8bf*)(qcs + 16) = lo8(qc1);
    *(v8bf*)(qcs + 24) = hi8(qc1);

    for (int kb = 0; kb < 16; ++kb) {
      const bf16_t* krow = sK + (kb * 16 + l4) * DH;
      v16bf k0 = ld16(krow + b_off, krow + b_off + 16);
      v16bf k1 = ld16(krow + 32 + b_off, krow + 32 + b_off + 16);
      v8f dacc = vzero8();
      dacc = wmma_bf16(qc0, k0, dacc);
      dacc = wmma_bf16(qc1, k1, dacc);
#pragma unroll
      for (int r = 0; r < 8; ++r) {
        float v = dacc[r];
        float tmax = v;
#pragma unroll
        for (int off = 1; off < 16; off <<= 1)
          tmax = fmaxf(tmax, __shfl_xor(tmax, off, 32));
        float mn = fmaxf(m[r], tmax);
        float e = __expf(v - mn);
#pragma unroll
        for (int off = 1; off < 16; off <<= 1) e += __shfl_xor(e, off, 32);
        s[r] = s[r] * __expf(m[r] - mn) + e;
        m[r] = mn;
      }
    }
  }
  __syncthreads();
  float inv_s[8];
#pragma unroll
  for (int r = 0; r < 8; ++r) inv_s[r] = 1.0f / s[r];

  // ================= pass 2a: emit attn, coalesced through LDS =============
  float* attnbase = attnp + ((size_t)bh * LA + qbase) * (LB * LC);
  for (int kb = 0; kb < 16; ++kb) {
    const bf16_t* krow = sK + (kb * 16 + l4) * DH;
    v16bf k0 = ld16(krow + b_off, krow + b_off + 16);
    v16bf k1 = ld16(krow + 32 + b_off, krow + 32 + b_off + 16);
    for (int c = 0; c < LC; ++c) {
      const bf16_t* qcs = sQc + c * 1024 + lane * 32;
      v16bf qc0 = ld16(qcs, qcs + 8);
      v16bf qc1 = ld16(qcs + 16, qcs + 24);
      v8f dacc = vzero8();
      dacc = wmma_bf16(qc0, k0, dacc);
      dacc = wmma_bf16(qc1, k1, dacc);
#pragma unroll
      for (int r = 0; r < 8; ++r) {
        float a = __expf(dacc[r] - m[r]) * inv_s[r];
        sA32[((r + hi * 8) * 16 + l4) * 32 + c] = a;  // [q][k16][c]
      }
    }
    __syncthreads();
    for (int i = lane; i < (16 * 512) / 4; i += 32) {
      int q = i >> 7;
      int off = (i & 127) * 4;
      float4 vv = *(const float4*)&sA32[q * 512 + off];
      *(float4*)(attnbase + (size_t)q * (LB * LC) + kb * 512 + off) = vv;
    }
    __syncthreads();
  }

  // ================= pass 2b: out[q,d] = sum_c C[c,d] * (attn_c @ V) ========
  v8f o0 = vzero8(), o1 = vzero8(), o2 = vzero8(), o3 = vzero8();
  for (int c = 0; c < LC; ++c) {
    const bf16_t* qcs = sQc + c * 1024 + lane * 32;
    v16bf qc0 = ld16(qcs, qcs + 8);
    v16bf qc1 = ld16(qcs + 16, qcs + 24);
    v8f a0 = vzero8(), a1 = vzero8(), a2 = vzero8(), a3 = vzero8();
    for (int kb2 = 0; kb2 < 8; ++kb2) {  // 32 k per iteration
#pragma unroll
      for (int sub = 0; sub < 2; ++sub) {
        const int kb = kb2 * 2 + sub;
        const bf16_t* krow = sK + (kb * 16 + l4) * DH;
        v16bf k0 = ld16(krow + b_off, krow + b_off + 16);
        v16bf k1 = ld16(krow + 32 + b_off, krow + 32 + b_off + 16);
        v8f dacc = vzero8();
        dacc = wmma_bf16(qc0, k0, dacc);
        dacc = wmma_bf16(qc1, k1, dacc);
#pragma unroll
        for (int r = 0; r < 8; ++r) {
          float a = __expf(dacc[r] - m[r]) * inv_s[r];
          sAb[(r + hi * 8) * 32 + sub * 16 + l4] = (bf16_t)a;  // [q][k32]
        }
      }
      __syncthreads();
      const bf16_t* arow = sAb + l4 * 32;
      v16bf afr = ld16(arow + a_off, arow + 16 + a_off);  // A: 16q x 32k
      const bf16_t* v0p = gvt + (size_t)(0 + l4) * LB + kb2 * 32 + b_off;
      const bf16_t* v1p = gvt + (size_t)(16 + l4) * LB + kb2 * 32 + b_off;
      const bf16_t* v2p = gvt + (size_t)(32 + l4) * LB + kb2 * 32 + b_off;
      const bf16_t* v3p = gvt + (size_t)(48 + l4) * LB + kb2 * 32 + b_off;
      a0 = wmma_bf16(afr, ld16(v0p, v0p + 16), a0);
      a1 = wmma_bf16(afr, ld16(v1p, v1p + 16), a1);
      a2 = wmma_bf16(afr, ld16(v2p, v2p + 16), a2);
      a3 = wmma_bf16(afr, ld16(v3p, v3p + 16), a3);
      __syncthreads();
    }
    const float cv0 = (float)sC[c * DH + 0 + l4];
    const float cv1 = (float)sC[c * DH + 16 + l4];
    const float cv2 = (float)sC[c * DH + 32 + l4];
    const float cv3 = (float)sC[c * DH + 48 + l4];
#pragma unroll
    for (int r = 0; r < 8; ++r) {
      o0[r] += a0[r] * cv0;
      o1[r] += a1[r] * cv1;
      o2[r] += a2[r] * cv2;
      o3[r] += a3[r] * cv3;
    }
  }

  // out [B, La, 512] f32; column n = h*64 + dt*16 + l4
#pragma unroll
  for (int r = 0; r < 8; ++r) {
    const int q = qbase + r + hi * 8;
    float* orow = outp + ((size_t)(b * LA + q)) * DM + h * DH;
    orow[0 + l4] = o0[r];
    orow[16 + l4] = o1[r];
    orow[32 + l4] = o2[r];
    orow[48 + l4] = o3[r];
  }
}

// ---------------------------------------------------------------- launcher
extern "C" void kernel_launch(void* const* d_in, const int* in_sizes, int n_in,
                              void* d_out, int out_size, void* d_ws,
                              size_t ws_size, hipStream_t stream) {
  const float* Qin = (const float*)d_in[0];
  const float* Kin = (const float*)d_in[1];
  const float* Vin = (const float*)d_in[2];
  const float* Ctok = (const float*)d_in[3];
  const float* Wq = (const float*)d_in[4];
  const float* bq = (const float*)d_in[5];
  const float* Wk = (const float*)d_in[6];
  const float* bk = (const float*)d_in[7];
  const float* Wv = (const float*)d_in[8];
  const float* bv = (const float*)d_in[9];

  char* ws = (char*)d_ws;
  size_t o = 0;
  bf16_t* qin_b = (bf16_t*)(ws + o); o += (size_t)B_ * LA * DM * 2;
  bf16_t* kin_b = (bf16_t*)(ws + o); o += (size_t)B_ * LB * DM * 2;
  bf16_t* vin_b = (bf16_t*)(ws + o); o += (size_t)B_ * LB * DM * 2;
  bf16_t* wq_b  = (bf16_t*)(ws + o); o += (size_t)DM * DM * 2;
  bf16_t* wk_b  = (bf16_t*)(ws + o); o += (size_t)DM * DM * 2;
  bf16_t* wv_b  = (bf16_t*)(ws + o); o += (size_t)DM * DM * 2;
  bf16_t* qhp   = (bf16_t*)(ws + o); o += (size_t)B_ * NH * LA * DH * 2;
  bf16_t* khp   = (bf16_t*)(ws + o); o += (size_t)B_ * NH * LB * DH * 2;
  bf16_t* vtp   = (bf16_t*)(ws + o); o += (size_t)B_ * NH * DH * LB * 2;
  bf16_t* chp   = (bf16_t*)(ws + o); o += (size_t)B_ * NH * LC * DH * 2;

  k_cvt_bf16<<<512, 256, 0, stream>>>(Qin, qin_b, B_ * LA * DM);
  k_cvt_bf16<<<512, 256, 0, stream>>>(Kin, kin_b, B_ * LB * DM);
  k_cvt_bf16<<<512, 256, 0, stream>>>(Vin, vin_b, B_ * LB * DM);
  k_cvt_bf16<<<256, 256, 0, stream>>>(Wq, wq_b, DM * DM);
  k_cvt_bf16<<<256, 256, 0, stream>>>(Wk, wk_b, DM * DM);
  k_cvt_bf16<<<256, 256, 0, stream>>>(Wv, wv_b, DM * DM);
  k_pack_c<<<(B_ * NH * LC * DH) / 256, 256, 0, stream>>>(Ctok, chp);

  k_proj<<<dim3((B_ * LA) / 16, DM / 16, 3), 32, 0, stream>>>(
      qin_b, kin_b, vin_b, wq_b, wk_b, wv_b, bq, bk, bv, qhp, khp, vtp);

  float* outp = (float*)d_out;
  float* attnp = outp + (size_t)B_ * LA * DM;
  k_tri_attn<<<dim3(LA / 16, B_ * NH), 32, 0, stream>>>(qhp, khp, vtp, chp,
                                                        outp, attnp);
}
